// DyGeo_GAT_Module_26414048870528
// MI455X (gfx1250) — matrix operand order repaired and checked by reading, not compiled, not run on hardware
//
#include <hip/hip_runtime.h>

typedef __attribute__((ext_vector_type(16))) _Float16 v16h;
typedef __attribute__((ext_vector_type(8)))  _Float16 v8h;
typedef __attribute__((ext_vector_type(8)))  float    v8f;

constexpr int Bsz = 4, S = 256, D = 512, H = 8, C = 64;
constexpr float NEG_SLOPE = 0.2f;
constexpr float LN_EPS = 1e-5f;

// ---------------- f16 fragment loaders (contiguous k order: k = half*16 + j) ----------------
__device__ __forceinline__ v16h cat8(v8h lo, v8h hi) {
  return __builtin_shufflevector(lo, hi, 0, 1, 2, 3, 4, 5, 6, 7,
                                 8, 9, 10, 11, 12, 13, 14, 15);
}

// A: 16x32 tile from row-major f16, tile origin `a`, leading dim lda.
// Per lane the 16 halves are contiguous: k=half*16+[0..16).
__device__ __forceinline__ v16h load_a_h(const _Float16* a, int lda, int lane) {
  int m = lane & 15, half = (lane >> 4) & 1;
  const _Float16* row = a + (long)m * lda + half * 16;
  v8h lo = *(const v8h*)(row);
  v8h hi = *(const v8h*)(row + 8);
  return cat8(lo, hi);
}

// B = M^T, M row-major NxK f16: lane n holds K=half*16+[0..16) contiguous -> one 32B load.
__device__ __forceinline__ v16h load_bT_h(const _Float16* b, int ldb, int lane) {
  int n = lane & 15, half = (lane >> 4) & 1;
  return *(const v16h*)(b + (long)n * ldb + half * 16);
}

// Pre-packed fragment order: lane's 16 halves contiguous.
__device__ __forceinline__ v16h load_pack_h(const _Float16* p) {
  return *(const v16h*)p;
}

__device__ __forceinline__ float wred_sum(float v) {
#pragma unroll
  for (int m = 16; m >= 1; m >>= 1) v += __shfl_xor(v, m, 32);
  return v;
}
__device__ __forceinline__ float wred_max(float v) {
#pragma unroll
  for (int m = 16; m >= 1; m >>= 1) v = fmaxf(v, __shfl_xor(v, m, 32));
  return v;
}

// ---------------- pack kernels ----------------
__global__ __launch_bounds__(256) void pack_f16_kernel(const float* __restrict__ src,
                                                       _Float16* __restrict__ dst, int n) {
  int i = blockIdx.x * 256 + threadIdx.x;
  if (i < n) dst[i] = (_Float16)src[i];
}

// W [512][512] row-major f32 -> fragment-ordered f16:
// Wp[((kt*32 + nt)*32 + lane)*16 + j], element j of lane = W[kt*32 + half*16 + j][nt*16 + (lane&15)]
__global__ __launch_bounds__(256) void pack_w_kernel(const float* __restrict__ W,
                                                     _Float16* __restrict__ Wp) {
  int t = blockIdx.x * 256 + threadIdx.x;  // 16*32*32 = 16384
  int lane = t & 31;
  int nt = (t >> 5) & 31;
  int kt = t >> 10;
  int n = nt * 16 + (lane & 15);
  int half = (lane >> 4) & 1;
  v16h out;
#pragma unroll
  for (int j = 0; j < 16; ++j) {
    int k = kt * 32 + half * 16 + j;
    out[j] = (_Float16)W[(long)k * D + n];
  }
  *(v16h*)(Wp + (long)t * 16) = out;
}

// XL [b][r][h*C+c] f32 -> XLt [b][h][c][r] f16 (so agg B-fragments are contiguous)
__global__ __launch_bounds__(256) void pack_xlt_kernel(const float* __restrict__ XL,
                                                       _Float16* __restrict__ XLt) {
  long i = (long)blockIdx.x * 256 + threadIdx.x;  // ((b*H+h)*C+c)*S + r
  int r = (int)(i & 255);
  int c = (int)((i >> 8) & 63);
  int h = (int)((i >> 14) & 7);
  int b = (int)(i >> 17);
  XLt[i] = (_Float16)XL[((long)(b * S + r)) * D + h * C + c];
}

// ---------------- Kernel 1: projections, wave = 16x64 strip ----------------
__global__ __launch_bounds__(256) void proj_gemm_kernel(const _Float16* __restrict__ Ah,
                                                        const _Float16* __restrict__ Wp,
                                                        const float* __restrict__ bias,
                                                        float* __restrict__ Yf,
                                                        _Float16* __restrict__ Yh) {
  int wave = blockIdx.x * 8 + (threadIdx.x >> 5);  // 512 waves: 64 tm x 8 strips
  int lane = threadIdx.x & 31;
  int tm = wave >> 3;
  int ts = wave & 7;  // strip: nt = ts*4 .. ts*4+3
  v8f acc[4] = {};
  const _Float16* abase = Ah + (long)tm * 16 * D;
  for (int kt = 0; kt < 16; ++kt) {
    if (kt + 1 < 16) __builtin_prefetch(abase + (kt + 1) * 32, 0, 0);
    v16h av = load_a_h(abase + kt * 32, D, lane);
    const _Float16* wp = Wp + (((long)kt * 32 + ts * 4) * 32 + lane) * 16;
#pragma unroll
    for (int t4 = 0; t4 < 4; ++t4) {
      v16h bv = load_pack_h(wp + (long)t4 * 32 * 16);
      acc[t4] = __builtin_amdgcn_wmma_f32_16x16x32_f16(false, av, false, bv,
                                                       (short)0, acc[t4], false, false);
    }
  }
  int nl = lane & 15, half = (lane >> 4) & 1;
  int nbase = ts * 64;
#pragma unroll
  for (int t4 = 0; t4 < 4; ++t4) {
    float bb = bias[nbase + t4 * 16 + nl];
#pragma unroll
    for (int v = 0; v < 8; ++v) {
      int m = tm * 16 + half * 8 + v;
      long off = (long)m * D + nbase + t4 * 16 + nl;
      float val = acc[t4][v] + bb;
      if (Yf) Yf[off] = val;
      if (Yh) Yh[off] = (_Float16)val;
    }
  }
}

// ---------------- Kernel 2: batched Y[b] = scale * A[b] @ B[b]^T (M=N=256) -----
__global__ __launch_bounds__(256) void gemm_abT_kernel(const _Float16* __restrict__ A,
                                                       const _Float16* __restrict__ Bm,
                                                       float* __restrict__ Y,
                                                       int K, float scale,
                                                       long aBatch, long bBatch, long yBatch) {
  int b = blockIdx.y;
  const _Float16* Ab = A + (long)b * aBatch;
  const _Float16* Bb = Bm + (long)b * bBatch;
  float* Yb = Y + (long)b * yBatch;
  int wave = blockIdx.x * 8 + (threadIdx.x >> 5);  // 64 waves: 16 tm x 4 strips
  int lane = threadIdx.x & 31;
  int tm = wave >> 2;
  int ts = wave & 3;  // nbase = ts*64
  int KT = K >> 5;
  v8f acc[4] = {};
  const _Float16* abase = Ab + (long)tm * 16 * K;
  for (int kt = 0; kt < KT; ++kt) {
    if (kt + 1 < KT) __builtin_prefetch(abase + (kt + 1) * 32, 0, 0);
    v16h av = load_a_h(abase + kt * 32, K, lane);
#pragma unroll
    for (int t4 = 0; t4 < 4; ++t4) {
      v16h bv = load_bT_h(Bb + (long)(ts * 64 + t4 * 16) * K + kt * 32, K, lane);
      acc[t4] = __builtin_amdgcn_wmma_f32_16x16x32_f16(false, av, false, bv,
                                                       (short)0, acc[t4], false, false);
    }
  }
  int nl = lane & 15, half = (lane >> 4) & 1;
#pragma unroll
  for (int t4 = 0; t4 < 4; ++t4) {
#pragma unroll
    for (int v = 0; v < 8; ++v) {
      int m = tm * 16 + half * 8 + v;
      Yb[(long)m * S + ts * 64 + t4 * 16 + nl] = scale * acc[t4][v];
    }
  }
}

// ---------------- Kernel 3: per-row squared norms of x ----------------
__global__ __launch_bounds__(256) void rowsq_kernel(const float* __restrict__ x,
                                                    float* __restrict__ sq) {
  int row = blockIdx.x * 8 + (threadIdx.x >> 5);
  int lane = threadIdx.x & 31;
  const float* r = x + (long)row * D;
  float s = 0.f;
#pragma unroll
  for (int i = 0; i < 16; ++i) { float v = r[lane + 32 * i]; s += v * v; }
  s = wred_sum(s);
  if (lane == 0) sq[row] = s;
}

// ---------------- Kernel 4: dist/sim + edge MLP 2->32->8 ----------------
__global__ __launch_bounds__(256) void edge_mlp_kernel(const float* __restrict__ gram,
                                                       const float* __restrict__ sq,
                                                       const float* __restrict__ We1,
                                                       const float* __restrict__ be1,
                                                       const float* __restrict__ We2,
                                                       const float* __restrict__ be2,
                                                       float* __restrict__ e) {
  long idx = (long)blockIdx.x * 256 + threadIdx.x;  // b*S*S + i*S + j
  int j = (int)(idx & 255);
  int i = (int)((idx >> 8) & 255);
  int b = (int)(idx >> 16);
  float g  = gram[idx];
  float si = sq[b * S + i], sj = sq[b * S + j];
  float d2 = fmaxf(si + sj - 2.f * g, 0.f);
  float dist = sqrtf(d2);
  float ni = fmaxf(sqrtf(si), 1e-12f);
  float nj = fmaxf(sqrtf(sj), 1e-12f);
  float sim = g / (ni * nj);
  float eh[8];
#pragma unroll
  for (int h = 0; h < 8; ++h) eh[h] = be2[h];
  for (int u = 0; u < 32; ++u) {
    float hu = fmaf(dist, We1[u], fmaf(sim, We1[32 + u], be1[u]));
    hu = fmaxf(hu, 0.f);
#pragma unroll
    for (int h = 0; h < 8; ++h) eh[h] = fmaf(hu, We2[u * 8 + h], eh[h]);
  }
  float4* op = (float4*)(e + idx * 8);
  op[0] = make_float4(eh[0], eh[1], eh[2], eh[3]);
  op[1] = make_float4(eh[4], eh[5], eh[6], eh[7]);
}

// ---------------- Kernel 5: GATv2 scores (nonlinear contraction, VALU) --------
// score layout: [b][t][h][r]
__global__ __launch_bounds__(256) void score_kernel(const float* __restrict__ XL,
                                                    const float* __restrict__ XR,
                                                    const float* __restrict__ e,
                                                    const float* __restrict__ Wedge,
                                                    const float* __restrict__ att,
                                                    float* __restrict__ score) {
  __shared__ float xr_s[D];
  __shared__ float att_s[D];
  __shared__ float wedge_s[H * D];
  int tid = threadIdx.x;
  long blk = blockIdx.x;  // b*2048 + t*8 + rc
  int rc = (int)(blk & 7);
  int t  = (int)((blk >> 3) & 255);
  int b  = (int)(blk >> 11);
  const float* xr_g = XR + ((long)b * S + t) * D;
  for (int i = tid; i < D; i += 256) { xr_s[i] = xr_g[i]; att_s[i] = att[i]; }
  for (int i = tid; i < H * D; i += 256) wedge_s[i] = Wedge[i];
  __syncthreads();

  int h  = tid >> 5;  // wave-uniform head -> all LDS reads are broadcasts
  int rl = tid & 31;
  int r  = rc * 32 + rl;
  const float4* e4 = (const float4*)(e + ((long)((b * S + r) * S) + t) * 8);
  float4 ea = e4[0], eb = e4[1];
  float e8[8] = {ea.x, ea.y, ea.z, ea.w, eb.x, eb.y, eb.z, eb.w};
  const float* xlp = XL + ((long)b * S + r) * D + h * C;
  const float* xrp = xr_s + h * C;
  const float* ap  = att_s + h * C;
  float acc = 0.f;
#pragma unroll 4
  for (int c = 0; c < C; ++c) {
    float ep = 0.f;
#pragma unroll
    for (int hp = 0; hp < 8; ++hp) ep = fmaf(e8[hp], wedge_s[hp * D + h * C + c], ep);
    float mv = xlp[c] + xrp[c] + ep;
    mv = mv > 0.f ? mv : NEG_SLOPE * mv;
    acc = fmaf(mv, ap[c], acc);
  }
  score[((long)(b * S + t) * H + h) * S + r] = acc;
}

// ---------------- Kernel 6: softmax over r, emit f16 alpha ----------------
__global__ __launch_bounds__(256) void softmax_kernel(const float* __restrict__ score,
                                                      _Float16* __restrict__ alphah) {
  long bt = blockIdx.x;
  int h = threadIdx.x >> 5, lane = threadIdx.x & 31;
  const float* row = score + (bt * H + h) * S;
  _Float16* orow = alphah + (bt * H + h) * S;
  float v[8];
  float mx = -3.4e38f;
#pragma unroll
  for (int i = 0; i < 8; ++i) { v[i] = row[lane + 32 * i]; mx = fmaxf(mx, v[i]); }
  mx = wred_max(mx);
  float s = 0.f;
#pragma unroll
  for (int i = 0; i < 8; ++i) { v[i] = expf(v[i] - mx); s += v[i]; }
  s = wred_sum(s);
  float inv = 1.f / s;
#pragma unroll
  for (int i = 0; i < 8; ++i) orow[lane + 32 * i] = (_Float16)(v[i] * inv);
}

// ---------------- Kernel 7: gat[b,t,h,:] = alpha[b,t,h,:] @ XL[b,:,h,:] -------
// alphah: [b][t][h][r] f16 (A, lda=H*S); XLt: [b][h][c][r] f16 (B^T form, ldb=S)
__global__ __launch_bounds__(256) void agg_gemm_kernel(const _Float16* __restrict__ alphah,
                                                       const _Float16* __restrict__ XLt,
                                                       float* __restrict__ gat) {
  int bh = blockIdx.y;
  int b = bh >> 3, h = bh & 7;
  const _Float16* Ab = alphah + (long)b * S * H * S + (long)h * S;
  const _Float16* Bb = XLt + ((long)(b * H + h)) * C * S;
  float* Yb = gat + (long)b * S * D + h * C;
  int wave = blockIdx.x * 8 + (threadIdx.x >> 5);  // 16 waves = tm; full N=64 strip
  int lane = threadIdx.x & 31;
  int tm = wave;
  v8f acc[4] = {};
  const _Float16* abase = Ab + (long)tm * 16 * (H * S);
  for (int kt = 0; kt < 8; ++kt) {  // K = S = 256
    if (kt + 1 < 8) __builtin_prefetch(abase + (kt + 1) * 32, 0, 0);
    v16h av = load_a_h(abase + kt * 32, H * S, lane);
#pragma unroll
    for (int t4 = 0; t4 < 4; ++t4) {
      v16h bv = load_bT_h(Bb + (long)(t4 * 16) * S + kt * 32, S, lane);
      acc[t4] = __builtin_amdgcn_wmma_f32_16x16x32_f16(false, av, false, bv,
                                                       (short)0, acc[t4], false, false);
    }
  }
  int nl = lane & 15, half = (lane >> 4) & 1;
#pragma unroll
  for (int t4 = 0; t4 < 4; ++t4) {
#pragma unroll
    for (int v = 0; v < 8; ++v) {
      int m = tm * 16 + half * 8 + v;
      Yb[(long)m * D + t4 * 16 + nl] = acc[t4][v];
    }
  }
}

// ---------------- Kernel 8: residual + bias + LayerNorm ----------------
__global__ __launch_bounds__(256) void ln_kernel(const float* __restrict__ x,
                                                 const float* __restrict__ gat,
                                                 const float* __restrict__ bias_gat,
                                                 const float* __restrict__ gamma,
                                                 const float* __restrict__ beta,
                                                 float* __restrict__ out) {
  int row = blockIdx.x * 8 + (threadIdx.x >> 5);
  int lane = threadIdx.x & 31;
  const float* xr = x + (long)row * D;
  const float* gr = gat + (long)row * D;
  float y[16];
  float s = 0.f, ss = 0.f;
#pragma unroll
  for (int i = 0; i < 16; ++i) {
    int d = lane + 32 * i;
    float v = xr[d] + gr[d] + bias_gat[d];
    y[i] = v; s += v; ss += v * v;
  }
  s = wred_sum(s); ss = wred_sum(ss);
  float mu = s * (1.f / D);
  float var = ss * (1.f / D) - mu * mu;
  float inv = rsqrtf(var + LN_EPS);
#pragma unroll
  for (int i = 0; i < 16; ++i) {
    int d = lane + 32 * i;
    out[(long)row * D + d] = (y[i] - mu) * inv * gamma[d] + beta[d];
  }
}

// ---------------- launch ----------------
extern "C" void kernel_launch(void* const* d_in, const int* in_sizes, int n_in,
                              void* d_out, int out_size, void* d_ws, size_t ws_size,
                              hipStream_t stream) {
  (void)in_sizes; (void)n_in; (void)out_size; (void)ws_size;
  const float* x        = (const float*)d_in[0];
  const float* Wq       = (const float*)d_in[1];
  const float* bq       = (const float*)d_in[2];
  const float* Wk       = (const float*)d_in[3];
  const float* bk       = (const float*)d_in[4];
  const float* We1      = (const float*)d_in[5];
  const float* be1      = (const float*)d_in[6];
  const float* We2      = (const float*)d_in[7];
  const float* be2      = (const float*)d_in[8];
  const float* Wl       = (const float*)d_in[9];
  const float* bl       = (const float*)d_in[10];
  const float* Wr       = (const float*)d_in[11];
  const float* br       = (const float*)d_in[12];
  const float* Wedge    = (const float*)d_in[13];
  const float* att      = (const float*)d_in[14];
  const float* bias_gat = (const float*)d_in[15];
  const float* gamma    = (const float*)d_in[16];
  const float* beta     = (const float*)d_in[17];

  char* cur = (char*)d_ws;
  auto alloc = [&](size_t bytes) { char* p = cur; cur += (bytes + 255) & ~(size_t)255; return p; };

  const long ND = (long)Bsz * S * D;     // 524288
  const long NSS = (long)Bsz * S * S;    // 262144
  _Float16* xh     = (_Float16*)alloc(ND * 2);
  _Float16* Wp0    = (_Float16*)alloc((size_t)D * D * 2);
  _Float16* Wp1    = (_Float16*)alloc((size_t)D * D * 2);
  _Float16* Wp2    = (_Float16*)alloc((size_t)D * D * 2);
  _Float16* Wp3    = (_Float16*)alloc((size_t)D * D * 2);
  _Float16* Qh     = (_Float16*)alloc(ND * 2);
  _Float16* Kh     = (_Float16*)alloc(ND * 2);
  float*    XL     = (float*)   alloc(ND * 4);
  float*    XR     = (float*)   alloc(ND * 4);
  _Float16* XLt    = (_Float16*)alloc(ND * 2);
  float*    gram   = (float*)   alloc(NSS * 4);
  float*    sq     = (float*)   alloc((size_t)Bsz * S * 4);
  float*    e      = (float*)   alloc(NSS * H * 4);
  float*    score  = (float*)   alloc(NSS * H * 4);
  _Float16* alphah = (_Float16*)alloc(NSS * H * 2);
  float*    gat    = (float*)   alloc(ND * 4);

  float* out = (float*)d_out;            // [B,S,D]
  float* adj = out + ND;                 // [B,S,S]

  dim3 blk(256);
  // operand packing
  pack_f16_kernel<<<2048, blk, 0, stream>>>(x, xh, (int)ND);
  pack_w_kernel<<<64, blk, 0, stream>>>(Wq, Wp0);
  pack_w_kernel<<<64, blk, 0, stream>>>(Wk, Wp1);
  pack_w_kernel<<<64, blk, 0, stream>>>(Wl, Wp2);
  pack_w_kernel<<<64, blk, 0, stream>>>(Wr, Wp3);
  // projections (Q,K -> f16 only; XL,XR -> f32)
  proj_gemm_kernel<<<64, blk, 0, stream>>>(xh, Wp0, bq, nullptr, Qh);
  proj_gemm_kernel<<<64, blk, 0, stream>>>(xh, Wp1, bk, nullptr, Kh);
  proj_gemm_kernel<<<64, blk, 0, stream>>>(xh, Wp2, bl, XL, nullptr);
  proj_gemm_kernel<<<64, blk, 0, stream>>>(xh, Wp3, br, XR, nullptr);
  // adjacency = Q K^T / (sqrt(C)*H) ; gram = x x^T
  gemm_abT_kernel<<<dim3(8, Bsz), blk, 0, stream>>>(Qh, Kh, adj, D, 1.f / 64.f,
                                                    (long)S * D, (long)S * D, (long)S * S);
  gemm_abT_kernel<<<dim3(8, Bsz), blk, 0, stream>>>(xh, xh, gram, D, 1.f,
                                                    (long)S * D, (long)S * D, (long)S * S);
  rowsq_kernel<<<128, blk, 0, stream>>>(x, sq);
  edge_mlp_kernel<<<1024, blk, 0, stream>>>(gram, sq, We1, be1, We2, be2, e);
  score_kernel<<<8192, blk, 0, stream>>>(XL, XR, e, Wedge, att, score);
  softmax_kernel<<<1024, blk, 0, stream>>>(score, alphah);
  pack_xlt_kernel<<<2048, blk, 0, stream>>>(XL, XLt);
  agg_gemm_kernel<<<dim3(2, Bsz * H), blk, 0, stream>>>(alphah, XLt, gat);
  ln_kernel<<<128, blk, 0, stream>>>(x, gat, bias_gat, gamma, beta, out);
}
